// WanSelfAttention_36481452212236
// MI455X (gfx1250) — matrix operand order repaired and checked by reading, not compiled, not run on hardware
//
#include <hip/hip_runtime.h>

// ---------------------------------------------------------------------------
// WanSelfAttention for MI455X (gfx1250, wave32, WMMA bf16 -> f32 accum,
// async global->LDS double-buffered GEMM)
// B=1, S=4096, DIM=1536, NH=12, HD=128
// ---------------------------------------------------------------------------

#define S_C   4096
#define DIM_C 1536
#define NH_C  12
#define HD_C  128
#define KSTEP 64

typedef __attribute__((ext_vector_type(16))) __bf16 v16bf;
typedef __attribute__((ext_vector_type(8)))  __bf16 v8bf;
typedef __attribute__((ext_vector_type(8)))  float  v8f;

// ---------------------------------------------------------------------------
// Fragment loaders (ISA 7.12.2 layouts, wave32):
//  A (16x32 bf16): lane L (L&15 = row), half = L>>4.
//    elements j=0..7  : k = 8*half + j           (contiguous 16B)
//    elements j=8..15 : k = 16 + 8*half + (j-8)  (contiguous 16B)
//  B (32x16 bf16): lane L (L&15 = col n), half = L>>4.
//    same k order as A (two contiguous 16B runs)
//  C/D (16x16 f32): element r -> row = r + 8*half, col = lane&15.
// ---------------------------------------------------------------------------

__device__ __forceinline__ v16bf load_a_frag(const __bf16* __restrict__ base,
                                             int ld, int row0, int k0, int lane) {
  const __bf16* p = base + (size_t)(row0 + (lane & 15)) * ld + k0 + ((lane >> 4) << 3);
  v8bf lo = *(const v8bf*)(p);
  v8bf hi = *(const v8bf*)(p + 16);
  v16bf a;
#pragma unroll
  for (int j = 0; j < 8; ++j) { a[j] = lo[j]; a[j + 8] = hi[j]; }
  return a;
}

// B[k, n] where column n is K-contiguous in memory with leading dim `ld`:
// element j = base[(n0 + lane&15)*ld + k0 + 8*half + j] (j<8), +16 for j>=8
__device__ __forceinline__ v16bf load_b_frag(const __bf16* __restrict__ base,
                                             size_t ld, int n0, int k0, int lane) {
  const __bf16* p = base + (size_t)(n0 + (lane & 15)) * ld + k0 + ((lane >> 4) << 3);
  v8bf lo = *(const v8bf*)(p);
  v8bf hi = *(const v8bf*)(p + 16);
  v16bf b;
#pragma unroll
  for (int j = 0; j < 8; ++j) { b[j] = lo[j]; b[j + 8] = hi[j]; }
  return b;
}

// Async DMA: 16 bytes global -> LDS, tracked by ASYNCcnt (CDNA5 async ops).
// The LDS address operand is derived from the real shared-memory pointer:
// low 32 bits of a generic LDS pointer are the addrspace(3) offset, and the
// ptrtoint escape makes the "memory" clobber cover the __shared__ array so
// the compiler keeps (and orders) the subsequent ds_load fragment reads.
__device__ __forceinline__ void async_copy_b128(void* lds_ptr, const void* gaddr) {
  unsigned lds_off = (unsigned)(unsigned long long)lds_ptr;
  unsigned long long ga = (unsigned long long)gaddr;
  asm volatile("global_load_async_to_lds_b128 %0, %1, off"
               :: "v"(lds_off), "v"(ga) : "memory");
}

// ---------------------------------------------------------------------------
// Elementwise f32 -> bf16 conversion
// ---------------------------------------------------------------------------
__global__ void cvt_f32_bf16(const float* __restrict__ in, __bf16* __restrict__ out, int n) {
  int i = blockIdx.x * blockDim.x + threadIdx.x;
  int stride = gridDim.x * blockDim.x;
  for (; i < n; i += stride) out[i] = (__bf16)in[i];
}

// ---------------------------------------------------------------------------
// GEMM: C[M,N] (f32) = A[M,K] (bf16) * W[N,K]^T (bf16) + bias[N]
// Block: 256 thr = 8 waves; block tile 128(M) x 64(N); wave tile 16 x 64.
// A/B tiles staged in LDS via async global->LDS copies, double-buffered:
// copy of tile t+1 overlaps WMMA on tile t; one s_wait_asynccnt per tile.
// ---------------------------------------------------------------------------
__global__ __launch_bounds__(256) void gemm_bf16_kernel(
    const __bf16* __restrict__ A, const __bf16* __restrict__ W,
    const float* __restrict__ bias, float* __restrict__ C,
    int M, int N, int K) {
  const int tid  = threadIdx.x;
  const int lane = tid & 31;
  const int wv   = tid >> 5;               // 0..7
  const int row0 = blockIdx.y * 128;
  const int col0 = blockIdx.x * 64;
  const int col  = lane & 15;
  const int half = lane >> 4;

  // [A0 16KB][A1 16KB][B0 8KB][B1 8KB] = 48KB (WGP LDS budget: 320KB)
  __shared__ __align__(16) char smem[2 * (128 * KSTEP + 64 * KSTEP) * 2];
  const unsigned aBufSz = 128 * KSTEP * 2;  // 16384
  const unsigned bBufSz = 64 * KSTEP * 2;   //  8192

  // 6 async b128 instructions per wave per tile (4 for A, 2 for B)
  auto copy_tile = [&](int buf, int k0) {
    char* aBase = smem + (unsigned)buf * aBufSz;
    char* bBase = smem + 2u * aBufSz + (unsigned)buf * bBufSz;
#pragma unroll
    for (int i = 0; i < 4; ++i) {
      int id = tid + 256 * i;             // 0..1023 : 128 rows x 8 segs
      int row = id >> 3, seg = id & 7;
      async_copy_b128(aBase + row * 128 + seg * 16,
                      A + (size_t)(row0 + row) * K + k0 + seg * 8);
    }
#pragma unroll
    for (int i = 0; i < 2; ++i) {
      int id = tid + 256 * i;             // 0..511 : 64 rows x 8 segs
      int row = id >> 3, seg = id & 7;
      async_copy_b128(bBase + row * 128 + seg * 16,
                      W + (size_t)(col0 + row) * K + k0 + seg * 8);
    }
  };

  v8f acc[4] = {};
  const int ntiles = K / KSTEP;
  copy_tile(0, 0);
  for (int t = 0; t < ntiles; ++t) {
    if (t + 1 < ntiles) {
      copy_tile((t + 1) & 1, (t + 1) * KSTEP);
      // next tile (6 copies) in flight; wait only for tile t's 6 copies
      asm volatile("s_wait_asynccnt 0x6" ::: "memory");
    } else {
      asm volatile("s_wait_asynccnt 0x0" ::: "memory");
    }
    __syncthreads();
    const __bf16* As = (const __bf16*)(smem + (t & 1) * aBufSz);
    const __bf16* Bs = (const __bf16*)(smem + 2 * aBufSz + (t & 1) * bBufSz);
#pragma unroll
    for (int ks = 0; ks < KSTEP; ks += 32) {
      v16bf a = load_a_frag(As, KSTEP, wv * 16, ks, lane);
#pragma unroll
      for (int c = 0; c < 4; ++c) {
        v16bf b = load_b_frag(Bs, (size_t)KSTEP, c * 16, ks, lane);
        acc[c] = __builtin_amdgcn_wmma_f32_16x16x32_bf16(
            false, a, false, b, (short)0, acc[c], false, false);
      }
    }
    __syncthreads();  // protect buffer (t&1) until all waves done reading
  }

#pragma unroll
  for (int c = 0; c < 4; ++c) {
    const int n = col0 + c * 16 + col;
    const float bv = bias[n];
#pragma unroll
    for (int r = 0; r < 8; ++r) {
      const int m = row0 + wv * 16 + r + 8 * half;
      C[(size_t)m * N + n] = acc[c][r] + bv;
    }
  }
}

// ---------------------------------------------------------------------------
// Fused RMSNorm (over DIM) + RoPE (interleaved pairs) for Q,K; V -> transposed
// bf16 [DIM][S]. Softmax scale folded into Q. One block per sequence position.
// ---------------------------------------------------------------------------
__global__ __launch_bounds__(256) void norm_rope_kernel(
    const float* __restrict__ qf, const float* __restrict__ kf,
    const float* __restrict__ vf,
    const float* __restrict__ nqw, const float* __restrict__ nkw,
    const float* __restrict__ freqs,           // [1024, 64]
    const int* __restrict__ p_gh, const int* __restrict__ p_gw,
    __bf16* __restrict__ qb, __bf16* __restrict__ kb, __bf16* __restrict__ vT) {
  const int s    = blockIdx.x;
  const int tid  = threadIdx.x;
  const int lane = tid & 31;
  const int wv   = tid >> 5;

  const float* qrow = qf + (size_t)s * DIM_C;
  const float* krow = kf + (size_t)s * DIM_C;
  const float* vrow = vf + (size_t)s * DIM_C;

  float sq = 0.f, sk = 0.f;
  for (int i = tid; i < DIM_C; i += 256) {
    float a = qrow[i]; sq += a * a;
    float b = krow[i]; sk += b * b;
  }
#pragma unroll
  for (int off = 16; off; off >>= 1) {
    sq += __shfl_xor(sq, off, 32);
    sk += __shfl_xor(sk, off, 32);
  }
  __shared__ float red[2][8];
  if (lane == 0) { red[0][wv] = sq; red[1][wv] = sk; }
  __syncthreads();
  float tq = 0.f, tk = 0.f;
#pragma unroll
  for (int w2 = 0; w2 < 8; ++w2) { tq += red[0][w2]; tk += red[1][w2]; }
  const float rq = rsqrtf(tq * (1.0f / DIM_C) + 1e-6f);
  const float rk = rsqrtf(tk * (1.0f / DIM_C) + 1e-6f);

  const int gh = *p_gh, gw = *p_gw;
  const int f = s / (gh * gw);
  const int h = (s / gw) % gh;
  const int w = s % gw;

  const float sm_scale = 0.08838834764831845f;  // 1/sqrt(HD=128)
  const int c3 = HD_C / 2;                       // 64
  const int s1 = c3 - 2 * (c3 / 3);              // 22
  const int s12 = s1 + c3 / 3;                   // 43

  for (int i = tid; i < DIM_C / 2; i += 256) {
    const int head = i >> 6;       // pair index / 64
    const int p = i & 63;
    const int pos = (p < s1) ? f : ((p < s12) ? h : w);
    const float ang = freqs[pos * 64 + p];
    const float ca = __cosf(ang), sa = __sinf(ang);
    const int e = head * HD_C + 2 * p;
    const size_t o = (size_t)s * DIM_C + e;
    // q: norm * weight, rotate, fold in softmax scale
    float xr = qrow[e] * rq * nqw[e];
    float xi = qrow[e + 1] * rq * nqw[e + 1];
    qb[o]     = (__bf16)((xr * ca - xi * sa) * sm_scale);
    qb[o + 1] = (__bf16)((xr * sa + xi * ca) * sm_scale);
    // k
    float yr = krow[e] * rk * nkw[e];
    float yi = krow[e + 1] * rk * nkw[e + 1];
    kb[o]     = (__bf16)(yr * ca - yi * sa);
    kb[o + 1] = (__bf16)(yr * sa + yi * ca);
  }
  // V transposed: vT[d][s], d = flat head*HD + hd (K-contiguous columns for WMMA B)
  for (int i = tid; i < DIM_C; i += 256)
    vT[(size_t)i * S_C + s] = (__bf16)vrow[i];
}

// ---------------------------------------------------------------------------
// Flash attention: block = 128 thr = 4 waves; grid = (S/64, NH).
// Each wave: 16 queries x HD=128, online softmax over key blocks of 32.
// Scores via 8 WMMA (2 key sub-tiles x 4 K-chunks), PV via 8 WMMA.
// P relayout (C-layout -> A-layout) through per-wave LDS.
// ---------------------------------------------------------------------------
__global__ __launch_bounds__(128) void attn_kernel(
    const __bf16* __restrict__ qb, const __bf16* __restrict__ kbm,
    const __bf16* __restrict__ vT, const int* __restrict__ seq_lens,
    __bf16* __restrict__ ob) {
  const int lane = threadIdx.x & 31;
  const int wv   = threadIdx.x >> 5;   // 0..3
  const int head = blockIdx.y;
  const int q0   = blockIdx.x * 64 + wv * 16;
  const int col  = lane & 15;
  const int half = lane >> 4;
  const int sl   = seq_lens[0];

  // Q fragments: 4 chunks of K=32 covering HD=128 (scale pre-folded)
  v16bf qfrag[4];
#pragma unroll
  for (int c = 0; c < 4; ++c)
    qfrag[c] = load_a_frag(qb + head * HD_C, DIM_C, q0, c * 32, lane);

  v8f o[8] = {};
  float mrow[8], lrow[8];
#pragma unroll
  for (int r = 0; r < 8; ++r) { mrow[r] = -1e30f; lrow[r] = 0.f; }

  __shared__ __align__(16) __bf16 plds[4][16 * 32];

  for (int kb0 = 0; kb0 < S_C; kb0 += 32) {
    // prefetch next key-block row for this lane into WGP-level cache
    if (kb0 + 32 < S_C)
      __builtin_prefetch(kbm + (size_t)(kb0 + 32 + col) * DIM_C + head * HD_C, 0, 3);

    // scores: two 16x16 tiles (keys kb0..+15 and kb0+16..+31)
    v8f sA = {}, sB = {};
#pragma unroll
    for (int c = 0; c < 4; ++c) {
      v16bf kfA = load_b_frag(kbm + head * HD_C, (size_t)DIM_C, kb0, c * 32, lane);
      sA = __builtin_amdgcn_wmma_f32_16x16x32_bf16(
          false, qfrag[c], false, kfA, (short)0, sA, false, false);
      v16bf kfB = load_b_frag(kbm + head * HD_C, (size_t)DIM_C, kb0 + 16, c * 32, lane);
      sB = __builtin_amdgcn_wmma_f32_16x16x32_bf16(
          false, qfrag[c], false, kfB, (short)0, sB, false, false);
    }

    const bool okA = (kb0 + col) < sl;
    const bool okB = (kb0 + 16 + col) < sl;

#pragma unroll
    for (int r = 0; r < 8; ++r) {
      float a = okA ? sA[r] : -1e30f;
      float b = okB ? sB[r] : -1e30f;
      float mx = fmaxf(a, b);
#pragma unroll
      for (int off = 8; off; off >>= 1) mx = fmaxf(mx, __shfl_xor(mx, off, 32));
      const float mnew = fmaxf(mrow[r], mx);
      const float pa = __expf(a - mnew);
      const float pb = __expf(b - mnew);
      float rsum = pa + pb;
#pragma unroll
      for (int off = 8; off; off >>= 1) rsum += __shfl_xor(rsum, off, 32);
      const float corr = __expf(mrow[r] - mnew);
      lrow[r] = lrow[r] * corr + rsum;
      mrow[r] = mnew;
#pragma unroll
      for (int c = 0; c < 8; ++c) o[c][r] *= corr;
      const int row = r + 8 * half;
      plds[wv][row * 32 + col]      = (__bf16)pa;
      plds[wv][row * 32 + col + 16] = (__bf16)pb;
    }
    __syncthreads();

    // reload P as a 16x32 A-fragment
    v16bf pfrag;
    {
      const __bf16* pp = &plds[wv][(lane & 15) * 32 + 8 * half];
      v8bf lo = *(const v8bf*)pp;
      v8bf hi = *(const v8bf*)(pp + 16);
#pragma unroll
      for (int j = 0; j < 8; ++j) { pfrag[j] = lo[j]; pfrag[j + 8] = hi[j]; }
    }
    __syncthreads();

    // O += P * V  (V^T layout makes B columns K-contiguous over the key axis)
#pragma unroll
    for (int c = 0; c < 8; ++c) {
      v16bf vfrag = load_b_frag(vT, (size_t)S_C, head * HD_C + c * 16, kb0, lane);
      o[c] = __builtin_amdgcn_wmma_f32_16x16x32_bf16(
          false, pfrag, false, vfrag, (short)0, o[c], false, false);
    }
  }

  // epilogue: normalize and store bf16
#pragma unroll
  for (int c = 0; c < 8; ++c) {
#pragma unroll
    for (int r = 0; r < 8; ++r) {
      const int m = q0 + r + 8 * half;
      const float val = o[c][r] / lrow[r];
      ob[(size_t)m * DIM_C + head * HD_C + c * 16 + col] = (__bf16)val;
    }
  }
}

// ---------------------------------------------------------------------------
// Host launcher
// ---------------------------------------------------------------------------
extern "C" void kernel_launch(void* const* d_in, const int* in_sizes, int n_in,
                              void* d_out, int out_size, void* d_ws, size_t ws_size,
                              hipStream_t stream) {
  (void)in_sizes; (void)n_in; (void)out_size; (void)ws_size;
  const float* x     = (const float*)d_in[0];
  const int*   slens = (const int*)d_in[1];
  const float* freqs = (const float*)d_in[2];
  const int*   gh    = (const int*)d_in[4];
  const int*   gw    = (const int*)d_in[5];
  const float* wq = (const float*)d_in[6];  const float* bq = (const float*)d_in[7];
  const float* wk = (const float*)d_in[8];  const float* bk = (const float*)d_in[9];
  const float* wvp = (const float*)d_in[10]; const float* bv = (const float*)d_in[11];
  const float* wo = (const float*)d_in[12]; const float* bo = (const float*)d_in[13];
  const float* nqw = (const float*)d_in[14]; const float* nkw = (const float*)d_in[15];
  float* out = (float*)d_out;

  char* ws = (char*)d_ws;
  size_t off = 0;
  auto take = [&](size_t bytes) -> void* {
    void* p = ws + off;
    off += (bytes + 255) & ~(size_t)255;
    return p;
  };
  const size_t SD = (size_t)S_C * DIM_C;
  const size_t WW = (size_t)DIM_C * DIM_C;
  __bf16* xb  = (__bf16*)take(SD * 2);
  __bf16* wqb = (__bf16*)take(WW * 2);
  __bf16* wkb = (__bf16*)take(WW * 2);
  __bf16* wvb = (__bf16*)take(WW * 2);
  __bf16* wob = (__bf16*)take(WW * 2);
  float*  qf  = (float*)take(SD * 4);
  float*  kf  = (float*)take(SD * 4);
  float*  vf  = (float*)take(SD * 4);
  __bf16* qb  = (__bf16*)take(SD * 2);
  __bf16* kbb = (__bf16*)take(SD * 2);
  __bf16* vT  = (__bf16*)take(SD * 2);
  __bf16* ob  = (__bf16*)take(SD * 2);

  cvt_f32_bf16<<<1024, 256, 0, stream>>>(x,   xb,  (int)SD);
  cvt_f32_bf16<<<1024, 256, 0, stream>>>(wq,  wqb, (int)WW);
  cvt_f32_bf16<<<1024, 256, 0, stream>>>(wk,  wkb, (int)WW);
  cvt_f32_bf16<<<1024, 256, 0, stream>>>(wvp, wvb, (int)WW);
  cvt_f32_bf16<<<1024, 256, 0, stream>>>(wo,  wob, (int)WW);

  dim3 gg(DIM_C / 64, S_C / 128);  // (24, 32)
  gemm_bf16_kernel<<<gg, 256, 0, stream>>>(xb, wqb, bq, qf, S_C, DIM_C, DIM_C);
  gemm_bf16_kernel<<<gg, 256, 0, stream>>>(xb, wkb, bk, kf, S_C, DIM_C, DIM_C);
  gemm_bf16_kernel<<<gg, 256, 0, stream>>>(xb, wvb, bv, vf, S_C, DIM_C, DIM_C);

  norm_rope_kernel<<<S_C, 256, 0, stream>>>(qf, kf, vf, nqw, nkw, freqs, gh, gw,
                                            qb, kbb, vT);

  dim3 ga(S_C / 64, NH_C);  // (64, 12)
  attn_kernel<<<ga, 128, 0, stream>>>(qb, kbb, vT, slens, ob);

  gemm_bf16_kernel<<<gg, 256, 0, stream>>>(ob, wob, bo, out, S_C, DIM_C, DIM_C);
}
